// DynamicLocalUpsampling_8572754723330
// MI455X (gfx1250) — matrix operand order repaired and checked by reading, not compiled, not run on hardware
//
#include <hip/hip_runtime.h>
#include <hip/hip_bf16.h>

// ---------------------------------------------------------------------------
// DynamicLocalUpsampling on gfx1250 (MI455X).
// conv3x3 stages implemented as im2col GEMMs on V_WMMA_F32_16X16X32_BF16,
// 4 independent accumulator chains per wave; branchless halo handling.
// Workspace layout (bytes):
//   [0        , 16MiB)   zA   : fp32 ping buffer (4 imgs x 1,048,576 f32)
//   [16MiB    , 32MiB)   zB   : fp32 pong buffer
//   [32MiB    , 64MiB)   U    : bf16 upsampled+skip tensor, HWC (4 x 4,194,304)
//   [64MiB    , +384KB)  W    : bf16 packed conv weights (B-fragment layout)
//   [..       , +128KB)  part : reduction partials (4 x 4096 x 2 f32)
//   [..       , +128B)   stats: mean/rstd per (stage,img)
// Total ~67.7 MB.
// ---------------------------------------------------------------------------

typedef __bf16 bf16;
typedef __attribute__((ext_vector_type(16))) __bf16 v16bf;
typedef __attribute__((ext_vector_type(8)))  __bf16 v8bf;
typedef __attribute__((ext_vector_type(8)))  float  v8f;

static constexpr size_t OFF_ZA = 0;
static constexpr size_t OFF_ZB = 16777216;
static constexpr size_t OFF_U  = 33554432;
static constexpr size_t OFF_W  = 67108864;
static constexpr size_t OFF_P  = 67502080;   // partials
static constexpr size_t OFF_S  = 67633152;   // stats

// packed-weight element offsets per stage (bf16 elements)
// layout: (((chunk*NTL + nt)*16 + ncol)*32 + slot); klocal = slot with bits 3 and 4 swapped
static constexpr int WOFF0 = 0;        // 72 chunks * 4 nt * 512 = 147456 elems
static constexpr int WOFF1 = 147456;   // 18 * 1 * 512           =   9216
static constexpr int WOFF2 = 156672;   //  5 * 1 * 512           =   2560
static constexpr int WOFF3 = 159232;   //  2 * 1 * 512           =   1024

__device__ __forceinline__ int iclamp(int v, int lo, int hi) {
    return v < lo ? lo : (v > hi ? hi : v);
}

// --------------------------- mask fill -------------------------------------
__global__ void fill_mask_kernel(float* __restrict__ out) {
    int idx = blockIdx.x * 256 + threadIdx.x;   // 4*1024*1024 elements
    out[idx] = -100.0f;
}

// ------------------- weight convert + pack (B fragment layout) --------------
template <int CIN, int COUT>
__global__ void cvt_weights_kernel(const float* __restrict__ cw, bf16* __restrict__ wd) {
    constexpr int NTL = (COUT + 15) / 16;
    int idx   = blockIdx.x * 256 + threadIdx.x;   // (KPAD/32)*NTL*512 total
    int kl    = (idx & 7) | ((idx & 8) << 1) | ((idx & 16) >> 1);
    int ncol  = (idx >> 5) & 15;
    int cn    = idx >> 9;                         // chunk*NTL + nt
    int nt    = cn % NTL;
    int chunk = cn / NTL;
    int n     = nt * 16 + ncol;
    int k     = chunk * 32 + kl;                  // k = tap*CIN + c
    float v = 0.0f;
    if (k < 9 * CIN && n < COUT) {
        int tap = k / CIN;          // CIN is a power of two -> shift
        int c   = k & (CIN - 1);
        int ty  = tap / 3, tx = tap % 3;
        v = cw[((n * CIN + c) * 3 + ty) * 3 + tx];
    }
    wd[idx] = (bf16)v;
}

// --------------------------- stats pass 1 ----------------------------------
// worst-case 1,048,576 elements per image at every stage -> 4096 blocks
template <int J, int C>
__global__ void reduce_kernel(const float* __restrict__ src, const int* __restrict__ bb,
                              float* __restrict__ part) {
    constexpr int SW = (J == 0) ? 64 : (64 << J);
    int img = blockIdx.y;
    int l = bb[img * 4 + 0], t = bb[img * 4 + 1];
    int r = bb[img * 4 + 2], b = bb[img * 4 + 3];
    int h = (b - t) << J, w = (r - l) << J;

    int idx = blockIdx.x * 256 + threadIdx.x;
    int c  = idx / (SW * SW);
    int rm = idx % (SW * SW);
    int y  = rm / SW;
    int x  = rm % SW;

    float v = 0.0f;
    if (c < C && y < h && x < w) {
        if (J == 0)
            v = src[((size_t)(img * C + c) * 64 + (t + y)) * 64 + (l + x)];
        else
            v = src[(size_t)img * 1048576 + ((size_t)c * SW + y) * SW + x];
    }

    __shared__ float ss[256], sq[256];
    int tid = threadIdx.x;
    ss[tid] = v; sq[tid] = v * v;
    __syncthreads();
    for (int s = 128; s > 0; s >>= 1) {
        if (tid < s) { ss[tid] += ss[tid + s]; sq[tid] += sq[tid + s]; }
        __syncthreads();
    }
    if (tid == 0) {
        part[((size_t)img * 4096 + blockIdx.x) * 2 + 0] = ss[0];
        part[((size_t)img * 4096 + blockIdx.x) * 2 + 1] = sq[0];
    }
}

// --------------------------- stats pass 2 (deterministic) -------------------
__global__ void finalize_stats_kernel(const float* __restrict__ part, const int* __restrict__ bb,
                                      float* __restrict__ stats, int J, int C) {
    int img = blockIdx.x;
    int tid = threadIdx.x;
    float a = 0.0f, q = 0.0f;
    for (int i = tid; i < 4096; i += 256) {       // fixed-order accumulation
        a += part[((size_t)img * 4096 + i) * 2 + 0];
        q += part[((size_t)img * 4096 + i) * 2 + 1];
    }
    __shared__ float ss[256], sq[256];
    ss[tid] = a; sq[tid] = q;
    __syncthreads();
    for (int s = 128; s > 0; s >>= 1) {
        if (tid < s) { ss[tid] += ss[tid + s]; sq[tid] += sq[tid + s]; }
        __syncthreads();
    }
    if (tid == 0) {
        int l = bb[img * 4 + 0], t = bb[img * 4 + 1];
        int r = bb[img * 4 + 2], b = bb[img * 4 + 3];
        float N    = (float)C * (float)((b - t) << J) * (float)((r - l) << J);
        float mean = ss[0] / N;
        float var  = sq[0] / N - mean * mean;
        stats[(J * 4 + img) * 2 + 0] = mean;
        stats[(J * 4 + img) * 2 + 1] = rsqrtf(var + 1e-5f);
    }
}

// ----------------- fused GroupNorm + bicubic2x + skip add -> bf16 HWC -------
// One thread produces a 2x2 output quad from a shared 5x5 source window
// (separable even/odd phases), then adds the skip tensor and stores bf16 HWC.
template <int J, int C>
__global__ void gnup_kernel(const float* __restrict__ src, const float* __restrict__ em,
                            const float* __restrict__ gwv, const float* __restrict__ gbv,
                            const float* __restrict__ stats, const int* __restrict__ bb,
                            bf16* __restrict__ U) {
    constexpr int UW = 128 << J;           // upsampled max dim == em plane dim
    constexpr int S  = 2 << J;             // em stride
    constexpr int SW = (J == 0) ? 64 : (64 << J);
    constexpr int HW = 64 << J;            // source max dim

    int img = blockIdx.y;
    int l = bb[img * 4 + 0], t = bb[img * 4 + 1];
    int r = bb[img * 4 + 2], b = bb[img * 4 + 3];
    int h = (b - t) << J, w = (r - l) << J;

    int idx = blockIdx.x * 256 + threadIdx.x;     // C*HW*HW = 1,048,576 / image
    int c   = idx % C;
    int pix = idx / C;
    int ix  = pix % HW;
    int iy  = pix / HW;
    if (iy >= h || ix >= w) return;

    float mean = stats[(J * 4 + img) * 2 + 0];
    float rstd = stats[(J * 4 + img) * 2 + 1];
    float ga = rstd * gwv[c];
    float gc = gbv[c] - mean * ga;

    const float Wc[4] = {-0.03515625f, 0.26171875f, 0.87890625f, -0.10546875f};

    float p[5][5];
#pragma unroll
    for (int dy = 0; dy < 5; dy++) {
        int yy = iclamp(iy + dy - 2, 0, h - 1);
#pragma unroll
        for (int dx = 0; dx < 5; dx++) {
            int xx = iclamp(ix + dx - 2, 0, w - 1);
            if (J == 0)
                p[dy][dx] = src[((size_t)(img * C + c) * 64 + (t + yy)) * 64 + (l + xx)];
            else
                p[dy][dx] = src[(size_t)img * 1048576 + ((size_t)c * SW + yy) * SW + xx];
        }
    }

    float he[5], ho[5];
#pragma unroll
    for (int i2 = 0; i2 < 5; i2++) {
        he[i2] = Wc[0]*p[i2][0] + Wc[1]*p[i2][1] + Wc[2]*p[i2][2] + Wc[3]*p[i2][3];
        ho[i2] = Wc[3]*p[i2][1] + Wc[2]*p[i2][2] + Wc[1]*p[i2][3] + Wc[0]*p[i2][4];
    }
    float o[2][2];
    o[0][0] = Wc[0]*he[0] + Wc[1]*he[1] + Wc[2]*he[2] + Wc[3]*he[3];
    o[0][1] = Wc[0]*ho[0] + Wc[1]*ho[1] + Wc[2]*ho[2] + Wc[3]*ho[3];
    o[1][0] = Wc[3]*he[1] + Wc[2]*he[2] + Wc[1]*he[3] + Wc[0]*he[4];
    o[1][1] = Wc[3]*ho[1] + Wc[2]*ho[2] + Wc[1]*ho[3] + Wc[0]*ho[4];

    bf16* Uo = U + (size_t)img * 4194304;
#pragma unroll
    for (int py = 0; py < 2; py++) {
#pragma unroll
        for (int px = 0; px < 2; px++) {
            int oy = 2 * iy + py, ox = 2 * ix + px;
            float val = o[py][px] * ga + gc
                      + em[(((size_t)img * C + c) * UW + (t * S + oy)) * UW + (l * S + ox)];
            Uo[((size_t)oy * UW + ox) * C + c] = (bf16)val;
        }
    }
}

// ----------------- conv3x3 as WMMA GEMM (bf16 in, f32 acc) ------------------
// Per wave: MT m-tiles (16 pixels each, along x) x NTL n-tiles (16 out-ch).
// MT*NTL == 4 independent accumulator chains. Halo handled branchlessly:
// loads are unconditional (addresses stay inside the workspace), then a
// scalar-condition vector select zeroes out-of-crop fragments.
template <int J, int CIN, int COUT, int MT>
__global__ __launch_bounds__(256)
void conv_gemm_kernel(const bf16* __restrict__ U, const bf16* __restrict__ W,
                      const float* __restrict__ bias, const int* __restrict__ bb,
                      float* __restrict__ dstZ, float* __restrict__ dmask) {
    constexpr int UW   = 128 << J;
    constexpr int KT   = 9 * CIN;
    constexpr int KPAD = ((KT + 31) / 32) * 32;
    constexpr int NTL  = (COUT + 15) / 16;
    constexpr int NCH  = KPAD / 32;
    constexpr int TXC  = UW / (16 * MT);

    int img = blockIdx.y;
    int l = bb[img * 4 + 0], t = bb[img * 4 + 1];
    int r = bb[img * 4 + 2], b = bb[img * 4 + 3];
    int uh = (b - t) << (J + 1);
    int uw = (r - l) << (J + 1);

    int wave = threadIdx.x >> 5;
    int lane = threadIdx.x & 31;
    int gwv  = blockIdx.x * 8 + wave;
    int tile_x = (gwv % TXC) * (16 * MT);
    int tile_y = gwv / TXC;
    if (tile_y >= uh || tile_x >= uw) return;     // wave-uniform early out

    const bf16* Ui = U + (size_t)img * 4194304;
    int m  = lane & 15;                           // A row / B col / D col
    int hi = (lane & 16) ? 1 : 0;                 // lane half selector
    const v16bf zfrag = {};
    const v8bf  zhalf = {};

    v8f acc[MT * NTL] = {};

    if constexpr (CIN % 32 == 0) {
        // 32-K chunk lies inside one tap; compile-time dy/dx per tap.
#pragma unroll
        for (int tap = 0; tap < 9; ++tap) {
            int y = tile_y + (tap / 3) - 1;
            int dx = (tap % 3) - 1;
            bool rowok = (unsigned)y < (unsigned)uh;
#pragma unroll 2
            for (int c0 = 0; c0 < CIN; c0 += 32) {
                int chunk = (tap * CIN + c0) >> 5;
                v16bf bfrag[NTL];
#pragma unroll
                for (int ni = 0; ni < NTL; ++ni)
                    bfrag[ni] = *(const v16bf*)(W + (((size_t)chunk * NTL + ni) * 16 + m) * 32
                                                  + (hi ? 16 : 0));
#pragma unroll
                for (int mi = 0; mi < MT; ++mi) {
                    int x = tile_x + mi * 16 + m + dx;
                    const bf16* p = Ui + ((size_t)y * UW + x) * CIN + c0 + (hi ? 8 : 0);
                    union { v16bf v; v8bf h[2]; } u;
                    u.h[0] = *(const v8bf*)p;          // K +0..7  (or +8..15)
                    u.h[1] = *(const v8bf*)(p + 16);   // K +16..23 (or +24..31)
                    bool ok = rowok && ((unsigned)x < (unsigned)uw);
                    v16bf afrag = ok ? u.v : zfrag;
#pragma unroll
                    for (int ni = 0; ni < NTL; ++ni)
                        acc[mi * NTL + ni] = __builtin_amdgcn_wmma_f32_16x16x32_bf16(
                            false, afrag, false, bfrag[ni], (short)0,
                            acc[mi * NTL + ni], false, false);
                }
            }
        }
    } else if constexpr (CIN == 16) {
        // 32-K chunk = two consecutive taps x 16 channels.
#pragma unroll
        for (int ch = 0; ch < NCH; ++ch) {
            v16bf bfrag = *(const v16bf*)(W + ((size_t)ch * 16 + m) * 32 + (hi ? 16 : 0));
            int t0 = ch * 2;                       // taps t0, t0+1
#pragma unroll
            for (int mi = 0; mi < MT; ++mi) {
                union { v16bf v; v8bf h[2]; } u;
#pragma unroll
                for (int hf = 0; hf < 2; ++hf) {
                    int tap = t0 + hf;
                    if (tap < 9) {
                        int y = tile_y + (tap / 3) - 1;
                        int x = tile_x + mi * 16 + m + (tap % 3) - 1;
                        v8bf ld = *(const v8bf*)(Ui + ((size_t)y * UW + x) * 16 + (hi ? 8 : 0));
                        bool ok = ((unsigned)y < (unsigned)uh) && ((unsigned)x < (unsigned)uw);
                        u.h[hf] = ok ? ld : zhalf;
                    } else {
                        u.h[hf] = zhalf;
                    }
                }
                acc[mi] = __builtin_amdgcn_wmma_f32_16x16x32_bf16(
                    false, u.v, false, bfrag, (short)0, acc[mi], false, false);
            }
        }
    } else {
        // CIN == 4 (tiny stage): element-wise gather.
#pragma unroll
        for (int ch = 0; ch < NCH; ++ch) {
            int kk = ch * 32;
            v16bf bfrag = *(const v16bf*)(W + ((size_t)ch * 16 + m) * 32 + (hi ? 16 : 0));
#pragma unroll
            for (int mi = 0; mi < MT; ++mi) {
                union { v16bf v; __bf16 e[16]; } u;
                u.v = zfrag;
#pragma unroll
                for (int e = 0; e < 16; e++) {
                    int k = kk + e + ((e >= 8) ? 8 : 0) + (hi ? 8 : 0);
                    if (k < KT) {
                        int tap = k / CIN;
                        int c   = k & (CIN - 1);
                        int y = tile_y + (tap / 3) - 1;
                        int x = tile_x + mi * 16 + m + (tap % 3) - 1;
                        if ((unsigned)y < (unsigned)uh && (unsigned)x < (unsigned)uw)
                            u.e[e] = Ui[((size_t)y * UW + x) * CIN + c];
                    }
                }
                acc[mi] = __builtin_amdgcn_wmma_f32_16x16x32_bf16(
                    false, u.v, false, bfrag, (short)0, acc[mi], false, false);
            }
        }
    }

    // D: lane n, row rI -> pixel = tile_x + mi*16 + rI + 8*hi, outch = nt*16 + (n&15)
#pragma unroll
    for (int mi = 0; mi < MT; ++mi) {
#pragma unroll
        for (int ni = 0; ni < NTL; ++ni) {
            int col = ni * 16 + m;
#pragma unroll
            for (int rI = 0; rI < 8; rI++) {
                int px = tile_x + mi * 16 + rI + (hi ? 8 : 0);
                if (px < uw && col < COUT) {
                    float v = acc[mi * NTL + ni][rI] + bias[col];
                    if constexpr (J < 3) {
                        v = v * (1.0f / (1.0f + __expf(-v)));          // SiLU
                        dstZ[(size_t)img * 1048576 + ((size_t)col * UW + tile_y) * UW + px] = v;
                    } else {
                        dmask[(size_t)img * 1048576
                              + (size_t)(t * 16 + tile_y) * 1024 + (l * 16 + px)] = v;
                    }
                }
            }
        }
    }
}

// ---------------------------------------------------------------------------
extern "C" void kernel_launch(void* const* d_in, const int* in_sizes, int n_in,
                              void* d_out, int out_size, void* d_ws, size_t ws_size,
                              hipStream_t stream) {
    (void)in_sizes; (void)n_in; (void)out_size; (void)ws_size;

    const float* x   = (const float*)d_in[0];
    const float* em0 = (const float*)d_in[1];
    const float* em1 = (const float*)d_in[2];
    const float* em2 = (const float*)d_in[3];
    const float* em3 = (const float*)d_in[4];
    const float* gw0 = (const float*)d_in[5];  const float* gb0 = (const float*)d_in[6];
    const float* cw0 = (const float*)d_in[7];  const float* cb0 = (const float*)d_in[8];
    const float* gw1 = (const float*)d_in[9];  const float* gb1 = (const float*)d_in[10];
    const float* cw1 = (const float*)d_in[11]; const float* cb1 = (const float*)d_in[12];
    const float* gw2 = (const float*)d_in[13]; const float* gb2 = (const float*)d_in[14];
    const float* cw2 = (const float*)d_in[15]; const float* cb2 = (const float*)d_in[16];
    const float* gw3 = (const float*)d_in[17]; const float* gb3 = (const float*)d_in[18];
    const float* cw3 = (const float*)d_in[19]; const float* cb3 = (const float*)d_in[20];
    const int*   bb  = (const int*)d_in[21];

    char*  ws    = (char*)d_ws;
    float* zA    = (float*)(ws + OFF_ZA);
    float* zB    = (float*)(ws + OFF_ZB);
    bf16*  U     = (bf16*)(ws + OFF_U);
    bf16*  Wall  = (bf16*)(ws + OFF_W);
    float* part  = (float*)(ws + OFF_P);
    float* stats = (float*)(ws + OFF_S);
    float* out   = (float*)d_out;

    bf16* W0 = Wall + WOFF0;
    bf16* W1 = Wall + WOFF1;
    bf16* W2 = Wall + WOFF2;
    bf16* W3 = Wall + WOFF3;

    // 0) fill mask + pack weights
    fill_mask_kernel<<<16384, 256, 0, stream>>>(out);
    cvt_weights_kernel<256, 64><<<576, 256, 0, stream>>>(cw0, W0);
    cvt_weights_kernel< 64, 16><<< 36, 256, 0, stream>>>(cw1, W1);
    cvt_weights_kernel< 16,  4><<< 10, 256, 0, stream>>>(cw2, W2);
    cvt_weights_kernel<  4,  1><<<  4, 256, 0, stream>>>(cw3, W3);

    dim3 rgrid(4096, 4), ggrid(4096, 4);

    // stage 0: x -> zA   (1 m-tile x 4 n-tiles per wave)
    reduce_kernel<0, 256><<<rgrid, 256, 0, stream>>>(x, bb, part);
    finalize_stats_kernel<<<4, 256, 0, stream>>>(part, bb, stats, 0, 256);
    gnup_kernel<0, 256><<<ggrid, 256, 0, stream>>>(x, em0, gw0, gb0, stats, bb, U);
    conv_gemm_kernel<0, 256, 64, 1><<<dim3(128, 4), 256, 0, stream>>>(U, W0, cb0, bb, zA, nullptr);

    // stage 1: zA -> zB  (4 m-tiles x 1 n-tile per wave)
    reduce_kernel<1, 64><<<rgrid, 256, 0, stream>>>(zA, bb, part);
    finalize_stats_kernel<<<4, 256, 0, stream>>>(part, bb, stats, 1, 64);
    gnup_kernel<1, 64><<<ggrid, 256, 0, stream>>>(zA, em1, gw1, gb1, stats, bb, U);
    conv_gemm_kernel<1, 64, 16, 4><<<dim3(128, 4), 256, 0, stream>>>(U, W1, cb1, bb, zB, nullptr);

    // stage 2: zB -> zA
    reduce_kernel<2, 16><<<rgrid, 256, 0, stream>>>(zB, bb, part);
    finalize_stats_kernel<<<4, 256, 0, stream>>>(part, bb, stats, 2, 16);
    gnup_kernel<2, 16><<<ggrid, 256, 0, stream>>>(zB, em2, gw2, gb2, stats, bb, U);
    conv_gemm_kernel<2, 16, 4, 4><<<dim3(512, 4), 256, 0, stream>>>(U, W2, cb2, bb, zA, nullptr);

    // stage 3: zA -> mask
    reduce_kernel<3, 4><<<rgrid, 256, 0, stream>>>(zA, bb, part);
    finalize_stats_kernel<<<4, 256, 0, stream>>>(part, bb, stats, 3, 4);
    gnup_kernel<3, 4><<<ggrid, 256, 0, stream>>>(zA, em3, gw3, gb3, stats, bb, U);
    conv_gemm_kernel<3, 4, 1, 4><<<dim3(2048, 4), 256, 0, stream>>>(U, W3, cb3, bb, nullptr, out);
}
